// _MOEAnomalyDetection_87986700026108
// MI455X (gfx1250) — matrix operand order repaired and checked by reading, not compiled, not run on hardware
//
#include <hip/hip_runtime.h>
#include <hip/hip_bf16.h>

// ---------------------------------------------------------------------------
// MoE anomaly detection, CDNA5 (gfx1250) wave32 + bf16 WMMA implementation.
//   D=512, E=16, H1=128, H2=64, R=32, N=65536
// Native __bf16 converts (v_cvt_pk_bf16_f32 class) replace software rounding.
// ---------------------------------------------------------------------------

#define D_IN  512
#define NEXP  16
#define H1    128
#define H2    64
#define RDIM  32
#define TM    128
#define TK    32
#define KA    (TK + 8)          // LDS row stride in bf16 elems; keeps 16B align

typedef __attribute__((ext_vector_type(16))) __bf16 v16bf;
typedef __attribute__((ext_vector_type(8)))  __bf16 v8bf;
typedef __attribute__((ext_vector_type(4)))  __bf16 v4bf;
typedef __attribute__((ext_vector_type(2)))  __bf16 v2bf;
typedef __attribute__((ext_vector_type(8)))  float  v8f;
typedef __attribute__((ext_vector_type(4)))  float  v4f;

struct FragPair { v8bf lo, hi; };

// Two 16B LDS chunks -> one v16bf WMMA operand (8 VGPRs), contiguous k order (k = 16h + s).
__device__ __forceinline__ v16bf load_frag(const __bf16* p0, const __bf16* p1) {
  FragPair fp;
  fp.lo = *(const v8bf*)p0;
  fp.hi = *(const v8bf*)p1;
  return __builtin_bit_cast(v16bf, fp);
}

// ---------------------------------------------------------------------------
__global__ void moe_zero_stats(float* p, int n) {
  int i = blockIdx.x * blockDim.x + threadIdx.x;
  if (i < n) p[i] = 0.f;
}

// ---------------------------------------------------------------------------
// K1: per-expert GEMM1 (M=N, K=512, N=H1) + fused gating on blockIdx.y==0.
__global__ void __launch_bounds__(256)
moe_gemm1(const float* __restrict__ x, const float* __restrict__ W1,
          const float* __restrict__ Wg, const float* __restrict__ bg,
          const float* __restrict__ ln_g, const float* __restrict__ ln_b,
          __bf16* __restrict__ h1, float* __restrict__ gate,
          float* __restrict__ sum1, float* __restrict__ sumsq1, int N) {
  __shared__ __bf16 As[2 * TM * KA];      // x tile, row-major [m][k]
  __shared__ __bf16 Bs[2 * H1 * KA];      // W1 tile, col-major [n][k]
  __shared__ __bf16 Gs[2 * NEXP * KA];    // Wg tile, col-major [e][k]
  __shared__ float gbuf[TM * NEXP];

  const int e      = blockIdx.y;
  const int m_base = blockIdx.x * TM;
  const int tid    = threadIdx.x;
  const int lane   = tid & 31;
  const int wave   = tid >> 5;

  const int wm = (wave >> 2) * 64;        // 2 M-blocks of 64 rows
  const int wn = (wave & 3) * 32;         // 4 N-blocks of 32 cols

  v8f z = {};
  v8f acc[4][2];
  v8f gacc[4];
  for (int mi = 0; mi < 4; ++mi) { acc[mi][0] = z; acc[mi][1] = z; gacc[mi] = z; }

  const float* xrow = x + (size_t)m_base * D_IN;
  const float* w1e  = W1 + (size_t)e * D_IN * H1;

  const int ar = lane & 15;
  const int ko = (lane >> 4) * 16;        // A: K-half select
  const int kh = (lane >> 4) * 16;        // B: K-half select
  const int cn = lane & 15;

  // stage thread mappings
  const int sm  = tid >> 1;               // A: row
  const int skh = (tid & 1) * 16;         // A: k half
  const int sn0 = (tid & 31) * 4;         // B: 4-column block
  const int skb = (tid >> 5) * 4;         // B: 4-k block

  auto stage_ab = [&](int kt, int buf) {
    const int k0 = kt * TK;
    __bf16* A = As + buf * (TM * KA);
    __bf16* B = Bs + buf * (H1 * KA);
    {   // x: 16 floats -> 16 bf16 (native packed converts)
      const float* src = xrow + (size_t)sm * D_IN + k0 + skh;
#pragma unroll
      for (int q = 0; q < 4; ++q) {
        v4f a = *(const v4f*)(src + 4 * q);
        *(v4bf*)(&A[sm * KA + skh + 4 * q]) = __builtin_convertvector(a, v4bf);
      }
    }
    {   // W1: 4x4 block, transpose to col-major
      const float* src = w1e + (size_t)(k0 + skb) * H1 + sn0;
      v4f r0 = *(const v4f*)(src);
      v4f r1 = *(const v4f*)(src + H1);
      v4f r2 = *(const v4f*)(src + 2 * H1);
      v4f r3 = *(const v4f*)(src + 3 * H1);
#pragma unroll
      for (int j = 0; j < 4; ++j) {
        v4bf p;
        p[0] = (__bf16)r0[j]; p[1] = (__bf16)r1[j];
        p[2] = (__bf16)r2[j]; p[3] = (__bf16)r3[j];
        *(v4bf*)(&B[(sn0 + j) * KA + skb]) = p;
      }
    }
  };
  auto stage_g = [&](int kt, int buf) {
    const int k0 = kt * TK;
    __bf16* G = Gs + buf * (NEXP * KA);
    for (int i = tid; i < TK * NEXP; i += 256) {
      int k = i >> 4, n = i & 15;
      G[n * KA + k] = (__bf16)Wg[(size_t)(k0 + k) * NEXP + n];
    }
  };
  auto mma_step = [&](int buf) {
    const __bf16* A = As + buf * (TM * KA);
    const __bf16* B = Bs + buf * (H1 * KA);
    v16bf bf0 = load_frag(&B[(wn + cn) * KA + kh],      &B[(wn + cn) * KA + kh + 8]);
    v16bf bf1 = load_frag(&B[(wn + 16 + cn) * KA + kh], &B[(wn + 16 + cn) * KA + kh + 8]);
#pragma unroll
    for (int mi = 0; mi < 4; ++mi) {
      const int row = wm + mi * 16 + ar;
      v16bf af = load_frag(&A[row * KA + ko], &A[row * KA + ko + 8]);
      acc[mi][0] = __builtin_amdgcn_wmma_f32_16x16x32_bf16(
          false, af, false, bf0, (short)0, acc[mi][0], false, false);
      acc[mi][1] = __builtin_amdgcn_wmma_f32_16x16x32_bf16(
          false, af, false, bf1, (short)0, acc[mi][1], false, false);
    }
  };
  auto mma_step_gate = [&](int buf) {
    const __bf16* A = As + buf * (TM * KA);
    const __bf16* B = Bs + buf * (H1 * KA);
    const __bf16* G = Gs + buf * (NEXP * KA);
    v16bf bf0 = load_frag(&B[(wn + cn) * KA + kh],      &B[(wn + cn) * KA + kh + 8]);
    v16bf bf1 = load_frag(&B[(wn + 16 + cn) * KA + kh], &B[(wn + 16 + cn) * KA + kh + 8]);
    v16bf gb  = load_frag(&G[cn * KA + kh],             &G[cn * KA + kh + 8]);
#pragma unroll
    for (int mi = 0; mi < 4; ++mi) {
      const int row = wm + mi * 16 + ar;
      v16bf af = load_frag(&A[row * KA + ko], &A[row * KA + ko + 8]);
      acc[mi][0] = __builtin_amdgcn_wmma_f32_16x16x32_bf16(
          false, af, false, bf0, (short)0, acc[mi][0], false, false);
      acc[mi][1] = __builtin_amdgcn_wmma_f32_16x16x32_bf16(
          false, af, false, bf1, (short)0, acc[mi][1], false, false);
      gacc[mi] = __builtin_amdgcn_wmma_f32_16x16x32_bf16(
          false, af, false, gb, (short)0, gacc[mi], false, false);
    }
  };

  constexpr int KT = D_IN / TK;           // 16 k-steps
  if (e == 0) {                           // block-uniform branch
    stage_ab(0, 0); stage_g(0, 0);
    __syncthreads();
    for (int kt = 0; kt < KT; ++kt) {
      const int cur = kt & 1;
      if (kt + 1 < KT) { stage_ab(kt + 1, cur ^ 1); stage_g(kt + 1, cur ^ 1); }
      mma_step_gate(cur);
      __syncthreads();
    }
  } else {
    stage_ab(0, 0);
    __syncthreads();
    for (int kt = 0; kt < KT; ++kt) {
      const int cur = kt & 1;
      if (kt + 1 < KT) stage_ab(kt + 1, cur ^ 1);
      mma_step(cur);
      __syncthreads();
    }
  }

  // epilogue: bf16 store + per-column batch stats
  const int rhalf = (lane >> 4) * 8;
  for (int ni = 0; ni < 2; ++ni) {
    const int col = wn + ni * 16 + cn;
    float csum = 0.f, csq = 0.f;
    for (int mi = 0; mi < 4; ++mi) {
      size_t gidx = (size_t)(m_base + wm + mi * 16 + rhalf) * (NEXP * H1)
                    + (size_t)e * H1 + col;
      v8f a = acc[mi][ni];
#pragma unroll
      for (int j = 0; j < 8; ++j) {
        float v = a[j];
        h1[gidx + (size_t)j * (NEXP * H1)] = (__bf16)v;
        csum += v; csq += v * v;
      }
    }
    csum += __shfl_xor(csum, 16, 32);
    csq  += __shfl_xor(csq,  16, 32);
    if (lane < 16) {
      atomicAdd(&sum1[e * H1 + col], csum);
      atomicAdd(&sumsq1[e * H1 + col], csq);
    }
  }

  if (e == 0) {                           // gating LN + softmax
    if ((wave & 3) == 0) {                // waves 0,4 hold rows 0-63/64-127
      for (int mi = 0; mi < 4; ++mi) {
        v8f ga = gacc[mi];
#pragma unroll
        for (int j = 0; j < 8; ++j)
          gbuf[(wm + mi * 16 + rhalf + j) * NEXP + cn] = ga[j];
      }
    }
    __syncthreads();
    if (tid < TM) {
      float g[NEXP]; float mu = 0.f;
#pragma unroll
      for (int t = 0; t < NEXP; ++t) { g[t] = gbuf[tid * NEXP + t] + bg[t]; mu += g[t]; }
      mu *= (1.f / NEXP);
      float var = 0.f;
#pragma unroll
      for (int t = 0; t < NEXP; ++t) { float d = g[t] - mu; var += d * d; }
      var *= (1.f / NEXP);
      float inv = rsqrtf(var + 1e-5f);
      float mx = -3.4e38f;
#pragma unroll
      for (int t = 0; t < NEXP; ++t) {
        g[t] = (g[t] - mu) * inv * ln_g[t] + ln_b[t];
        mx = fmaxf(mx, g[t]);
      }
      float s = 0.f;
#pragma unroll
      for (int t = 0; t < NEXP; ++t) { g[t] = __expf(g[t] - mx); s += g[t]; }
      float rs = 1.f / s;
      size_t gb_ = (size_t)(m_base + tid) * NEXP;
#pragma unroll
      for (int t = 0; t < NEXP; ++t) gate[gb_ + t] = g[t] * rs;
    }
  }
}

// ---------------------------------------------------------------------------
// stats -> (mean, 1/sqrt(var+eps))
__global__ void moe_bn_finalize(const float* __restrict__ sum,
                                const float* __restrict__ sumsq,
                                float* __restrict__ mean, float* __restrict__ inv,
                                int n, float invN) {
  int i = blockIdx.x * blockDim.x + threadIdx.x;
  if (i < n) {
    float mu  = sum[i] * invN;
    float var = sumsq[i] * invN - mu * mu;
    mean[i] = mu;
    inv[i]  = rsqrtf(fmaxf(var, 0.f) + 1e-4f);
  }
}

// v[e][k] = sum_r W3[e][k][r] * Ws[r]
__global__ void moe_fold_w3(const float* __restrict__ W3, const float* __restrict__ Ws,
                            float* __restrict__ v) {
  int i = blockIdx.x * blockDim.x + threadIdx.x;
  if (i < NEXP * H2) {
    const float* w = W3 + (size_t)i * RDIM;
    float s = 0.f;
#pragma unroll
    for (int r = 0; r < RDIM; ++r) s += w[r] * Ws[r];
    v[i] = s;
  }
}

// ---------------------------------------------------------------------------
// K2: per-expert GEMM2 (M=N, K=H1, N=H2); BN1+lrelu fused into A staging.
__global__ void __launch_bounds__(256)
moe_gemm2(const __bf16* __restrict__ h1, const float* __restrict__ W2,
          const float* __restrict__ mean1, const float* __restrict__ inv1,
          __bf16* __restrict__ h2,
          float* __restrict__ sum2, float* __restrict__ sumsq2, int N) {
  __shared__ __bf16 As[2 * TM * KA];
  __shared__ __bf16 Bs[2 * H2 * KA];
  __shared__ __attribute__((aligned(16))) float smean[H1];
  __shared__ __attribute__((aligned(16))) float sinv[H1];

  const int e      = blockIdx.y;
  const int m_base = blockIdx.x * TM;
  const int tid    = threadIdx.x;
  const int lane   = tid & 31;
  const int wave   = tid >> 5;

  const int wm = (wave >> 2) * 64;
  const int wn = (wave & 3) * 16;

  if (tid < H1) { smean[tid] = mean1[e * H1 + tid]; sinv[tid] = inv1[e * H1 + tid]; }
  __syncthreads();

  v8f z = {};
  v8f acc[4];
  for (int mi = 0; mi < 4; ++mi) acc[mi] = z;

  const float* w2e = W2 + (size_t)e * H1 * H2;
  const int ar = lane & 15;
  const int ko = (lane >> 4) * 16;
  const int kh = (lane >> 4) * 16;
  const int cn = lane & 15;

  const int sm  = tid >> 1;
  const int skh = (tid & 1) * 16;
  const int sn0 = (tid & 15) * 4;         // W2: 4-column block
  const int skb = (tid >> 4) * 2;         // W2: 2-k block

  auto stage = [&](int kt, int buf) {
    const int k0 = kt * TK;
    __bf16* A = As + buf * (TM * KA);
    __bf16* B = Bs + buf * (H2 * KA);
    {   // h1 row chunk: 16 bf16 (32B), BN1+lrelu, repack
      const int f0 = k0 + skh;
      const __bf16* src = h1 + (size_t)(m_base + sm) * (NEXP * H1) + e * H1 + f0;
      v8bf u0 = *(const v8bf*)(src);
      v8bf u1 = *(const v8bf*)(src + 8);
      v8f f0v = __builtin_convertvector(u0, v8f);
      v8f f1v = __builtin_convertvector(u1, v8f);
#pragma unroll
      for (int q = 0; q < 4; ++q) {
        v4f mn = *(const v4f*)(&smean[f0 + 4 * q]);
        v4f iv = *(const v4f*)(&sinv[f0 + 4 * q]);
        v4f hn;
#pragma unroll
        for (int j = 0; j < 4; ++j) {
          int t = 4 * q + j;
          float val = (t < 8) ? f0v[t & 7] : f1v[t & 7];
          float d = (val - mn[j]) * iv[j];
          hn[j] = d >= 0.f ? d : 0.01f * d;
        }
        *(v4bf*)(&A[sm * KA + skh + 4 * q]) = __builtin_convertvector(hn, v4bf);
      }
    }
    {   // W2: 4x2 block, transpose to col-major
      const float* src = w2e + (size_t)(k0 + skb) * H2 + sn0;
      v4f r0 = *(const v4f*)(src);
      v4f r1 = *(const v4f*)(src + H2);
#pragma unroll
      for (int j = 0; j < 4; ++j) {
        v2bf p;
        p[0] = (__bf16)r0[j]; p[1] = (__bf16)r1[j];
        *(v2bf*)(&B[(sn0 + j) * KA + skb]) = p;
      }
    }
  };

  constexpr int KT = H1 / TK;             // 4 k-steps
  stage(0, 0);
  __syncthreads();
  for (int kt = 0; kt < KT; ++kt) {
    const int cur = kt & 1;
    if (kt + 1 < KT) stage(kt + 1, cur ^ 1);
    const __bf16* A = As + cur * (TM * KA);
    const __bf16* B = Bs + cur * (H2 * KA);
    v16bf bf = load_frag(&B[(wn + cn) * KA + kh], &B[(wn + cn) * KA + kh + 8]);
#pragma unroll
    for (int mi = 0; mi < 4; ++mi) {
      const int row = wm + mi * 16 + ar;
      v16bf af = load_frag(&A[row * KA + ko], &A[row * KA + ko + 8]);
      acc[mi] = __builtin_amdgcn_wmma_f32_16x16x32_bf16(
          false, af, false, bf, (short)0, acc[mi], false, false);
    }
    __syncthreads();
  }

  const int rhalf = (lane >> 4) * 8;
  const int col = wn + cn;
  float csum = 0.f, csq = 0.f;
  for (int mi = 0; mi < 4; ++mi) {
    size_t gidx = (size_t)(m_base + wm + mi * 16 + rhalf) * (NEXP * H2)
                  + (size_t)e * H2 + col;
    v8f a = acc[mi];
#pragma unroll
    for (int j = 0; j < 8; ++j) {
      float v = a[j];
      h2[gidx + (size_t)j * (NEXP * H2)] = (__bf16)v;
      csum += v; csq += v * v;
    }
  }
  csum += __shfl_xor(csum, 16, 32);
  csq  += __shfl_xor(csq,  16, 32);
  if (lane < 16) {
    atomicAdd(&sum2[e * H2 + col], csum);
    atomicAdd(&sumsq2[e * H2 + col], csq);
  }
}

// ---------------------------------------------------------------------------
// K3: one wave per token. Per lane: two contiguous 16-elem chunks (32B vector
// loads), BN2+lrelu+dot folded into (mean, inv*v) tables, 7-shuffle reduce.
__global__ void __launch_bounds__(256)
moe_final(const __bf16* __restrict__ h2, const float* __restrict__ gate,
          const float* __restrict__ mean2, const float* __restrict__ inv2,
          const float* __restrict__ vfold, const float* __restrict__ bs,
          float* __restrict__ out, int N) {
  __shared__ __attribute__((aligned(16))) float smean[NEXP * H2];
  __shared__ __attribute__((aligned(16))) float siv[NEXP * H2];   // inv * v
  for (int i = threadIdx.x; i < NEXP * H2; i += 256) {
    smean[i] = mean2[i];
    siv[i]   = inv2[i] * vfold[i];
  }
  __syncthreads();
  const float bsv = bs[0];
  const int lane = threadIdx.x & 31;
  const int wid  = blockIdx.x * 8 + (threadIdx.x >> 5);
  const int nw   = gridDim.x * 8;

  for (int n = wid; n < N; n += nw) {
    const __bf16* row = h2 + (size_t)n * (NEXP * H2);
    float part[2];
#pragma unroll
    for (int c = 0; c < 2; ++c) {
      const int ci = lane + 32 * c;       // chunk index; expert = ci>>2
      const int base = ci * 16;
      v8bf u0 = *(const v8bf*)(row + base);
      v8bf u1 = *(const v8bf*)(row + base + 8);
      v8f f0v = __builtin_convertvector(u0, v8f);
      v8f f1v = __builtin_convertvector(u1, v8f);
      float a = 0.f;
#pragma unroll
      for (int q = 0; q < 4; ++q) {
        v4f mn = *(const v4f*)(&smean[base + 4 * q]);
        v4f iv = *(const v4f*)(&siv[base + 4 * q]);
#pragma unroll
        for (int j = 0; j < 4; ++j) {
          int t = 4 * q + j;
          float val = (t < 8) ? f0v[t & 7] : f1v[t & 7];
          float d = (val - mn[j]) * iv[j];  // (bn * v), sign(bn)==sign(val-mean)
          a += (val >= mn[j]) ? d : 0.01f * d;
        }
      }
      part[c] = a;
    }
    // reduce within each 4-lane expert group
    part[0] += __shfl_xor(part[0], 1, 32); part[0] += __shfl_xor(part[0], 2, 32);
    part[1] += __shfl_xor(part[1], 1, 32); part[1] += __shfl_xor(part[1], 2, 32);
    float s = 0.f;
    if ((lane & 3) == 0) {
      int g = lane >> 2;                  // experts g and g+8
      s = gate[(size_t)n * NEXP + g]     * (part[0] + bsv)
        + gate[(size_t)n * NEXP + 8 + g] * (part[1] + bsv);
    }
    s += __shfl_xor(s, 4, 32); s += __shfl_xor(s, 8, 32); s += __shfl_xor(s, 16, 32);
    if (lane == 0) out[n] = s;
  }
}

// ---------------------------------------------------------------------------
extern "C" void kernel_launch(void* const* d_in, const int* in_sizes, int n_in,
                              void* d_out, int out_size, void* d_ws, size_t ws_size,
                              hipStream_t stream) {
  const float* x    = (const float*)d_in[0];
  const float* Wg   = (const float*)d_in[1];
  const float* bg   = (const float*)d_in[2];
  const float* ln_g = (const float*)d_in[3];
  const float* ln_b = (const float*)d_in[4];
  const float* W1   = (const float*)d_in[5];
  const float* W2   = (const float*)d_in[6];
  const float* W3   = (const float*)d_in[7];
  const float* Ws   = (const float*)d_in[8];
  const float* bs   = (const float*)d_in[9];
  float* out = (float*)d_out;

  const int N = in_sizes[0] / D_IN;

  // workspace layout
  float* sum1   = (float*)d_ws;                  // 2048
  float* sumsq1 = sum1 + NEXP * H1;              // 2048
  float* sum2   = sumsq1 + NEXP * H1;            // 1024
  float* sumsq2 = sum2 + NEXP * H2;              // 1024
  float* mean1  = sumsq2 + NEXP * H2;            // 2048
  float* inv1   = mean1 + NEXP * H1;             // 2048
  float* mean2  = inv1 + NEXP * H1;              // 1024
  float* inv2   = mean2 + NEXP * H2;             // 1024
  float* vfold  = inv2 + NEXP * H2;              // 1024
  float* gate   = vfold + NEXP * H2;             // N*16
  __bf16* h1 = (__bf16*)(gate + (size_t)N * NEXP);
  __bf16* h2 = h1 + (size_t)N * NEXP * H1;

  const int nstats = NEXP * H1 * 2 + NEXP * H2 * 2;   // 6144 contiguous floats
  moe_zero_stats<<<(nstats + 255) / 256, 256, 0, stream>>>(sum1, nstats);

  dim3 g1(N / TM, NEXP);
  moe_gemm1<<<g1, 256, 0, stream>>>(x, W1, Wg, bg, ln_g, ln_b,
                                    h1, gate, sum1, sumsq1, N);
  moe_bn_finalize<<<(NEXP * H1 + 255) / 256, 256, 0, stream>>>(
      sum1, sumsq1, mean1, inv1, NEXP * H1, 1.f / (float)N);
  moe_fold_w3<<<(NEXP * H2 + 255) / 256, 256, 0, stream>>>(W3, Ws, vfold);

  dim3 g2(N / TM, NEXP);
  moe_gemm2<<<g2, 256, 0, stream>>>(h1, W2, mean1, inv1, h2, sum2, sumsq2, N);
  moe_bn_finalize<<<(NEXP * H2 + 255) / 256, 256, 0, stream>>>(
      sum2, sumsq2, mean2, inv2, NEXP * H2, 1.f / (float)N);

  moe_final<<<512, 256, 0, stream>>>(h2, gate, mean2, inv2, vfold, bs, out, N);
}
